// GIN_33062658245469
// MI455X (gfx1250) — matrix operand order repaired and checked by reading, not compiled, not run on hardware
//
#include <hip/hip_runtime.h>
#include <cmath>

typedef __attribute__((ext_vector_type(16))) _Float16 v16h;
typedef __attribute__((ext_vector_type(8)))  float    v8f;

#define DIMF   64
#define NGRAPH 128
#define NCLS   10
#define BN_EPS 1e-5f
#define GEMM_BLOCKS 128

// ---------------------------------------------------------------- utilities
__global__ void copy_f4(const float4* __restrict__ src, float4* __restrict__ dst, int n4) {
  int i = blockIdx.x * blockDim.x + threadIdx.x;
  if (i < n4) dst[i] = src[i];
}

__global__ void zero_f(float* __restrict__ p, int n) {
  int i = blockIdx.x * blockDim.x + threadIdx.x;
  if (i < n) p[i] = 0.0f;
}

// -------------------------------------------------- edge scatter-add (GIN agg)
__global__ void edge_scatter(const float* __restrict__ feat, const int* __restrict__ ei,
                             float* __restrict__ agg, int E, int F) {
  int nchunk = F >> 2;
  long long idx   = (long long)blockIdx.x * blockDim.x + threadIdx.x;
  long long total = (long long)E * nchunk;
  if (idx >= total) return;
  int e = (int)(idx / nchunk);
  int c = (int)(idx % nchunk) << 2;
  int src = ei[e];
  int dst = ei[E + e];
  const float4 v = *(const float4*)(feat + (size_t)src * F + c);
  float* o = agg + (size_t)dst * F + c;
  atomicAdd(o + 0, v.x);
  atomicAdd(o + 1, v.y);
  atomicAdd(o + 2, v.z);
  atomicAdd(o + 3, v.w);
}

// -------------------------------------------------- WMMA GEMM: out[N,64] = act(in[N,K] @ W[K,64] + b)
// Persistent: 8 waves/block grid-stride over 16-row M tiles. Weight panel
// pre-swizzled in LDS into B-fragment layout; for K=64 the 8 fragments are
// hoisted into VGPRs before the M loop (steady state has no LDS waits).
// Out-of-range rows are clamped (stores/stats masked) -> no EXEC divergence.
template <int K>
__global__ __launch_bounds__(256)
void gemm_wmma64(const float* __restrict__ in, const float* __restrict__ W,
                 const float* __restrict__ bias, float* __restrict__ out,
                 int nrows, int do_relu, float* __restrict__ stats) {
  constexpr int NCK = K / 32;                       // k-chunks
  __shared__ _Float16 sWf[NCK * 2048];              // [ck][tile][lane][16 halves]
  __shared__ float ssum[DIMF];
  __shared__ float ssq[DIMF];

  const int tid = threadIdx.x;
#pragma unroll
  for (int ii = 0; ii < NCK * 8; ++ii) {            // NCK*2048 / 256 threads
    const int idx = ii * 256 + tid;
    const int i  = idx & 15;
    const int l  = (idx >> 4) & 31;
    const int t  = (idx >> 9) & 3;
    const int ck = idx >> 11;
    const int k  = (ck << 5) + ((l & 16) >> 1) + (i & 7) + ((i & 8) << 1); // B layout: lane half -> K 0..7 / 8..15 (+16)
    const int n  = (t << 4) + (l & 15);
    sWf[idx] = (_Float16)W[k * DIMF + n];
  }
  if (stats && tid < DIMF) { ssum[tid] = 0.0f; ssq[tid] = 0.0f; }
  __syncthreads();

  const int lane  = tid & 31;
  const int wid   = tid >> 5;
  const int kbase = (lane & 16) >> 1;               // A layout: lane half -> K 0..7 / 8..15 (+16)
  const int ncol  = lane & 15;
  const int moff  = (lane < 16) ? 0 : 8;            // C/D layout
  const int numTiles   = (nrows + 15) >> 4;
  const int waveId     = blockIdx.x * 8 + wid;
  const int waveStride = gridDim.x * 8;

  float bv[4];
#pragma unroll
  for (int t = 0; t < 4; ++t) bv[t] = bias[t * 16 + ncol];

  // Hoist B fragments into registers for K=64 (8 * v16h = 64 VGPRs).
  v16h breg[(K == 64) ? NCK * 4 : 1];
  if constexpr (K == 64) {
#pragma unroll
    for (int ck = 0; ck < NCK; ++ck)
#pragma unroll
      for (int t = 0; t < 4; ++t)
        breg[ck * 4 + t] = *(const v16h*)(sWf + (ck << 11) + (t << 9) + (lane << 4));
  }

  for (int mt = waveId; mt < numTiles; mt += waveStride) {
    const int m0 = mt << 4;
    int mrow = m0 + (lane & 15);                    // A layout: lane -> M row
    if (mrow > nrows - 1) mrow = nrows - 1;         // clamp: masked at store/stats
    const float* rp0 = in + (size_t)mrow * K + kbase;
    v8f acc[4] = {};

#pragma unroll
    for (int ck = 0; ck < NCK; ++ck) {
      const float* rp = rp0 + (ck << 5);
      const float4 f0 = ((const float4*)rp)[0];
      const float4 f1 = ((const float4*)rp)[1];
      const float4 f2 = ((const float4*)(rp + 16))[0];
      const float4 f3 = ((const float4*)(rp + 16))[1];
      v16h a;
      a[0]  = (_Float16)f0.x; a[1]  = (_Float16)f0.y; a[2]  = (_Float16)f0.z; a[3]  = (_Float16)f0.w;
      a[4]  = (_Float16)f1.x; a[5]  = (_Float16)f1.y; a[6]  = (_Float16)f1.z; a[7]  = (_Float16)f1.w;
      a[8]  = (_Float16)f2.x; a[9]  = (_Float16)f2.y; a[10] = (_Float16)f2.z; a[11] = (_Float16)f2.w;
      a[12] = (_Float16)f3.x; a[13] = (_Float16)f3.y; a[14] = (_Float16)f3.z; a[15] = (_Float16)f3.w;

#pragma unroll
      for (int t = 0; t < 4; ++t) {
        v16h b;
        if constexpr (K == 64) {
          b = breg[ck * 4 + t];
        } else {
          b = *(const v16h*)(sWf + (ck << 11) + (t << 9) + (lane << 4));
        }
        acc[t] = __builtin_amdgcn_wmma_f32_16x16x32_f16(
            /*neg_a=*/false, a, /*neg_b=*/false, b,
            /*c_mod=*/(short)0, acc[t], /*reuse_a=*/false, /*reuse_b=*/false);
      }
    }

    // ---- epilogue: bias, ReLU, store, fused BN-stat reduction
    const bool full = (m0 + 16 <= nrows);
#pragma unroll
    for (int t = 0; t < 4; ++t) {
      const int n = t * 16 + ncol;
#pragma unroll
      for (int r = 0; r < 8; ++r) {
        const int m = m0 + r + moff;
        float v = acc[t][r] + bv[t];
        if (do_relu) v = fmaxf(v, 0.0f);
        if (full || m < nrows) {
          out[(size_t)m * DIMF + n] = v;
          if (stats) {
            atomicAdd(&ssum[n], v);
            atomicAdd(&ssq[n], v * v);
          }
        }
      }
    }
  }

  if (stats) {
    __syncthreads();
    if (tid < DIMF) {
      atomicAdd(&stats[tid], ssum[tid]);
      atomicAdd(&stats[DIMF + tid], ssq[tid]);
    }
  }
}

// -------------------------------------------------- BatchNorm apply (float4, training-mode stats)
// Optionally also (a) mirrors the normalized value into the next layer's
// aggregation base buffer (saves a separate copy pass), and (b) performs the
// global-add pool atomics for the final layer (saves a separate read pass).
__global__ void bn_apply(float4* __restrict__ h, const float* __restrict__ stats,
                         const float* __restrict__ gamma, const float* __restrict__ beta,
                         int nrows, float invN, float4* __restrict__ copy_dst,
                         const int* __restrict__ batch, float* __restrict__ pooled) {
  int idx = blockIdx.x * blockDim.x + threadIdx.x;
  if (idx >= nrows * (DIMF / 4)) return;
  int j4 = (idx & (DIMF / 4 - 1)) << 2;
  float4 v = h[idx];
  float* vp = &v.x;
#pragma unroll
  for (int q = 0; q < 4; ++q) {
    int j = j4 + q;
    float mean = stats[j] * invN;
    float var  = stats[DIMF + j] * invN - mean * mean;
    vp[q] = (vp[q] - mean) * rsqrtf(var + BN_EPS) * gamma[j] + beta[j];
  }
  h[idx] = v;
  if (copy_dst) copy_dst[idx] = v;
  if (pooled) {
    int row = idx >> 4;
    float* pp = pooled + (size_t)batch[row] * DIMF + j4;
    atomicAdd(pp + 0, v.x);
    atomicAdd(pp + 1, v.y);
    atomicAdd(pp + 2, v.z);
    atomicAdd(pp + 3, v.w);
  }
}

// -------------------------------------------------- head: fc1 (per (g,j)), fc2 + log_softmax (per g)
__global__ void head_fc1(const float* __restrict__ pooled, const float* __restrict__ w,
                         const float* __restrict__ b, float* __restrict__ z1) {
  int idx = blockIdx.x * blockDim.x + threadIdx.x;
  if (idx >= NGRAPH * DIMF) return;
  int g = idx >> 6;
  int j = idx & (DIMF - 1);
  float s = b[j];
  const float* p = pooled + (size_t)g * DIMF;
#pragma unroll 8
  for (int k = 0; k < DIMF; ++k) s += p[k] * w[k * DIMF + j];
  z1[idx] = fmaxf(s, 0.0f);
}

__global__ void head_fc2_lsm(const float* __restrict__ z1, const float* __restrict__ w,
                             const float* __restrict__ b, float* __restrict__ out) {
  int g = blockIdx.x * blockDim.x + threadIdx.x;
  if (g >= NGRAPH) return;
  const float* p = z1 + (size_t)g * DIMF;
  float z[NCLS];
#pragma unroll
  for (int c = 0; c < NCLS; ++c) z[c] = b[c];
  for (int k = 0; k < DIMF; ++k) {
    float pk = p[k];
#pragma unroll
    for (int c = 0; c < NCLS; ++c) z[c] += pk * w[k * NCLS + c];
  }
  float mx = z[0];
#pragma unroll
  for (int c = 1; c < NCLS; ++c) mx = fmaxf(mx, z[c]);
  float s = 0.0f;
#pragma unroll
  for (int c = 0; c < NCLS; ++c) s += __expf(z[c] - mx);
  float lse = __logf(s);
#pragma unroll
  for (int c = 0; c < NCLS; ++c) out[(size_t)g * NCLS + c] = z[c] - mx - lse;
}

// ================================================================ launcher
extern "C" void kernel_launch(void* const* d_in, const int* in_sizes, int n_in,
                              void* d_out, int out_size, void* d_ws, size_t ws_size,
                              hipStream_t stream) {
  (void)n_in; (void)out_size; (void)ws_size;

  const float* x      = (const float*)d_in[0];
  const int*   ei     = (const int*)d_in[1];
  const int*   batch  = (const int*)d_in[2];
  const float* W1a    = (const float*)d_in[3];
  const float* b1a    = (const float*)d_in[4];
  const float* W1b    = (const float*)d_in[5];
  const float* b1b    = (const float*)d_in[6];
  const float* Wa     = (const float*)d_in[7];
  const float* ba     = (const float*)d_in[8];
  const float* Wb     = (const float*)d_in[9];
  const float* bb     = (const float*)d_in[10];
  const float* gammas = (const float*)d_in[11];
  const float* betas  = (const float*)d_in[12];
  const float* fc1_w  = (const float*)d_in[13];
  const float* fc1_b  = (const float*)d_in[14];
  const float* fc2_w  = (const float*)d_in[15];
  const float* fc2_b  = (const float*)d_in[16];
  float* out = (float*)d_out;

  const int F_IN = 128;
  const int N = in_sizes[0] / F_IN;
  const int E = in_sizes[1] / 2;
  const float invN = 1.0f / (float)N;

  // workspace carve-up (floats)
  float* A      = (float*)d_ws;                      // [N,128] layer-1 (x + agg)
  float* Bm     = A + (size_t)N * F_IN;              // [N,64] MLP hidden
  float* Ccur   = Bm + (size_t)N * DIMF;             // [N,64] current h
  float* D      = Ccur + (size_t)N * DIMF;           // [N,64] agg for layers 2..5
  float* stats  = D + (size_t)N * DIMF;              // [128] col sum / sumsq
  float* pooled = stats + 2 * DIMF;                  // [128,64]
  float* z1     = pooled + (size_t)NGRAPH * DIMF;    // [128,64]

  const int TB = 256;
  const int numTiles = (N + 15) / 16;
  const int gemm_grid = (GEMM_BLOCKS * 8 < numTiles) ? GEMM_BLOCKS : (numTiles + 7) / 8;
  const int ne4 = N * DIMF / 4;
  const int np = NGRAPH * DIMF;

  // pooled must be zero before the final fused BN+pool pass
  zero_f<<<(np + TB - 1) / TB, TB, 0, stream>>>(pooled, np);

  // ---------------- layer 1: agg on x (F=128) ----------------
  {
    int n4 = N * F_IN / 4;
    copy_f4<<<(n4 + TB - 1) / TB, TB, 0, stream>>>((const float4*)x, (float4*)A, n4);
    long long sth = (long long)E * (F_IN / 4);
    edge_scatter<<<(int)((sth + TB - 1) / TB), TB, 0, stream>>>(x, ei, A, E, F_IN);
    gemm_wmma64<128><<<gemm_grid, TB, 0, stream>>>(A, W1a, b1a, Bm, N, 1, nullptr);
    zero_f<<<1, 2 * DIMF, 0, stream>>>(stats, 2 * DIMF);
    gemm_wmma64<64><<<gemm_grid, TB, 0, stream>>>(Bm, W1b, b1b, Ccur, N, 1, stats);
    // fused: normalized h also written into D (next layer's agg base)
    bn_apply<<<(ne4 + TB - 1) / TB, TB, 0, stream>>>(
        (float4*)Ccur, stats, gammas, betas, N, invN, (float4*)D, nullptr, nullptr);
  }

  // ---------------- layers 2..5: agg on h (F=64) ----------------
  for (int l = 0; l < 4; ++l) {
    long long sth = (long long)E * (DIMF / 4);
    edge_scatter<<<(int)((sth + TB - 1) / TB), TB, 0, stream>>>(Ccur, ei, D, E, DIMF);
    gemm_wmma64<64><<<gemm_grid, TB, 0, stream>>>(
        D, Wa + (size_t)l * DIMF * DIMF, ba + (size_t)l * DIMF, Bm, N, 1, nullptr);
    zero_f<<<1, 2 * DIMF, 0, stream>>>(stats, 2 * DIMF);
    gemm_wmma64<64><<<gemm_grid, TB, 0, stream>>>(
        Bm, Wb + (size_t)l * DIMF * DIMF, bb + (size_t)l * DIMF, Ccur, N, 1, stats);
    const bool last = (l == 3);
    // fused: layers 2..4 mirror into D; layer 5 performs the global-add pool
    bn_apply<<<(ne4 + TB - 1) / TB, TB, 0, stream>>>(
        (float4*)Ccur, stats, gammas + (size_t)(l + 1) * DIMF, betas + (size_t)(l + 1) * DIMF,
        N, invN, last ? nullptr : (float4*)D, last ? batch : nullptr, last ? pooled : nullptr);
  }

  // ---------------- head ----------------
  head_fc1<<<(np + TB - 1) / TB, TB, 0, stream>>>(pooled, fc1_w, fc1_b, z1);
  head_fc2_lsm<<<1, NGRAPH, 0, stream>>>(z1, fc2_w, fc2_b, out);
}
